// StreamingTransformerLite_45853070852388
// MI455X (gfx1250) — matrix operand order repaired and checked by reading, not compiled, not run on hardware
//
#include <hip/hip_runtime.h>
#include <hip/hip_bf16.h>

// ---------------------------------------------------------------------------
// StreamingTransformerLite for MI455X (gfx1250): bf16 WMMA GEMMs + fp32
// windowed attention. All matrix math via v_wmma_f32_16x16x32_bf16.
// ---------------------------------------------------------------------------

#define BATCH 4
#define TDN   1024          // tokens per batch
#define NTOK  4096          // BATCH * TDN
#define DMODEL 512
#define NHEAD 8
#define DKH   64
#define FFDIM 2048
#define WINSZ 256

typedef unsigned short u16;
typedef __attribute__((ext_vector_type(16))) __bf16 v16bf;
typedef __attribute__((ext_vector_type(8)))  float  v8f;

union FragU { v16bf v; uint4 u[2]; };

__device__ __forceinline__ float bf2f(u16 h) {
    union { unsigned int i; float f; } c; c.i = ((unsigned int)h) << 16; return c.f;
}
__device__ __forceinline__ u16 f2bf(float f) {
    union { float f; unsigned int i; } c; c.f = f;
    unsigned int u = c.i;
    u += 0x7fffu + ((u >> 16) & 1u);          // round-to-nearest-even
    return (u16)(u >> 16);
}
__device__ __forceinline__ float gelu_exact(float x) {
    return 0.5f * x * (1.0f + erff(x * 0.70710678118654752f));
}

// A fragment (16x32 bf16, row-major source, row stride K):
//   lane<16: row=lane,   elems 0..7 => K = k+0..7,   elems 8..15 => K = k+16..23
//   lane>=16: row=lane-16, elems 0..7 => K = k+8..15, elems 8..15 => K = k+24..31
__device__ __forceinline__ v16bf load_a_frag(const u16* rowp, int k, int half) {
    FragU f;
    const u16* p = rowp + k + half * 8;
    f.u[0] = *(const uint4*)(p);
    f.u[1] = *(const uint4*)(p + 16);
    return f.v;
}
// B fragment (32x16 bf16); B supplied transposed [N,K] row-major:
//   lane<16: col=lane, elems 0..7 => K = k+0..7, elems 8..15 => K = k+16..23 ; lane>=16: col=lane-16, elems 0..7 => K = k+8..15, elems 8..15 => K = k+24..31
__device__ __forceinline__ v16bf load_b_frag(const u16* colp, int k, int half) {
    FragU f;
    const u16* p = colp + k + half * 8;
    f.u[0] = *(const uint4*)(p);
    f.u[1] = *(const uint4*)(p + 16);
    return f.v;
}

// ---------------------------------------------------------------------------
// WMMA GEMM: out[M,N] = act(A[M,K] @ Wt[N,K]^T + bias) (+ resid)
// A, Wt in bf16. Each wave computes a 32x32 tile (2x2 WMMA tiles).
// ---------------------------------------------------------------------------
template<int HAS_BIAS, int DO_GELU, int ADD_RESID, int OUT_BF16>
__global__ __launch_bounds__(256)
void gemm_bf16_wmma(const u16* __restrict__ A, const u16* __restrict__ Wt,
                    const float* __restrict__ bias, const float* __restrict__ resid,
                    float* __restrict__ outF, u16* __restrict__ outH,
                    int M, int N, int K)
{
    int wave = (blockIdx.x * blockDim.x + threadIdx.x) >> 5;
    int lane = threadIdx.x & 31;
    int tilesN = N >> 5;
    int tm = wave / tilesN;
    int tn = wave - tm * tilesN;
    if (tm >= (M >> 5)) return;
    int m0 = tm << 5, n0 = tn << 5;
    int half = lane >> 4, l15 = lane & 15;

    const u16* a0p = A  + (size_t)(m0 + l15) * K;
    const u16* a1p = a0p + (size_t)16 * K;
    const u16* b0p = Wt + (size_t)(n0 + l15) * K;
    const u16* b1p = b0p + (size_t)16 * K;

    v8f acc00 = {}, acc01 = {}, acc10 = {}, acc11 = {};
    for (int k = 0; k < K; k += 32) {
        v16bf a0 = load_a_frag(a0p, k, half);
        v16bf a1 = load_a_frag(a1p, k, half);
        v16bf b0 = load_b_frag(b0p, k, half);
        v16bf b1 = load_b_frag(b1p, k, half);
        acc00 = __builtin_amdgcn_wmma_f32_16x16x32_bf16(false, a0, false, b0, (short)0, acc00, false, false);
        acc01 = __builtin_amdgcn_wmma_f32_16x16x32_bf16(false, a0, false, b1, (short)0, acc01, false, false);
        acc10 = __builtin_amdgcn_wmma_f32_16x16x32_bf16(false, a1, false, b0, (short)0, acc10, false, false);
        acc11 = __builtin_amdgcn_wmma_f32_16x16x32_bf16(false, a1, false, b1, (short)0, acc11, false, false);
    }

    #pragma unroll
    for (int sub = 0; sub < 4; sub++) {
        v8f acc = (sub == 0) ? acc00 : (sub == 1) ? acc01 : (sub == 2) ? acc10 : acc11;
        int mi = (sub >> 1) << 4;
        int ni = (sub & 1) << 4;
        int n = n0 + ni + l15;
        float bn = 0.0f;
        if (HAS_BIAS) bn = bias[n];
        #pragma unroll
        for (int r = 0; r < 8; r++) {
            int m = m0 + mi + r + half * 8;   // D layout: VGPR r -> row r (lanes 0-15) / r+8 (16-31)
            float x = acc[r] + bn;
            if (DO_GELU)   x = gelu_exact(x);
            if (ADD_RESID) x += resid[(size_t)m * N + n];
            if (OUT_BF16)  outH[(size_t)m * N + n] = f2bf(x);
            else           outF[(size_t)m * N + n] = x;
        }
    }
}

// ---------------------------------------------------------------------------
// Downsample conv + pad-mask: one thread per token.
// ---------------------------------------------------------------------------
__global__ void downsample_kernel(const float* __restrict__ x,
                                  const float* __restrict__ conv_w,
                                  const float* __restrict__ conv_b,
                                  float* __restrict__ xdown,
                                  float* __restrict__ kpm)
{
    int t = blockIdx.x * blockDim.x + threadIdx.x;
    if (t >= NTOK) return;
    const float* p = x + (size_t)t * 5;
    float s = conv_b[0];
    float pm = 0.0f;
    #pragma unroll
    for (int i = 0; i < 5; i++) {
        float v = p[i];
        s += v * conv_w[i];
        if (v == -1.0f) pm = 1.0f;
    }
    xdown[t] = s;
    kpm[t] = pm;
}

// h[tok,d] = xdown[tok]*in_w[d] + in_b[d] + pos[t,d]
__global__ void embed_kernel(const float* __restrict__ xdown,
                             const float* __restrict__ in_w,
                             const float* __restrict__ in_b,
                             const float* __restrict__ pos,
                             float* __restrict__ h)
{
    int i = blockIdx.x * blockDim.x + threadIdx.x;
    if (i >= NTOK * DMODEL) return;
    int tok = i >> 9;
    int d = i & (DMODEL - 1);
    int t = tok & (TDN - 1);
    h[i] = xdown[tok] * in_w[d] + in_b[d] + pos[(size_t)t * DMODEL + d];
}

// fp32 [K,N] -> bf16 transposed [N,K]
__global__ void convert_transpose_kernel(const float* __restrict__ src,
                                         u16* __restrict__ dst, int K, int N)
{
    int i = blockIdx.x * blockDim.x + threadIdx.x;
    if (i >= K * N) return;
    int k = i / N;
    int n = i - k * N;
    dst[(size_t)n * K + k] = f2bf(src[(size_t)k * N + n]);
}

// ---------------------------------------------------------------------------
// LayerNorm (wave per token, 512 = 16 floats/lane), bf16 output.
// ---------------------------------------------------------------------------
__global__ __launch_bounds__(256)
void layernorm_bf16(const float* __restrict__ x, const float* __restrict__ g,
                    const float* __restrict__ b, u16* __restrict__ out, int ntok)
{
    int w = threadIdx.x >> 5;
    int lane = threadIdx.x & 31;
    int tok = blockIdx.x * 8 + w;
    if (tok >= ntok) return;
    const float* row = x + (size_t)tok * DMODEL;
    int d0 = lane * 16;
    float4 v[4];
    float s = 0.0f, ss = 0.0f;
    #pragma unroll
    for (int i = 0; i < 4; i++) {
        v[i] = *(const float4*)(row + d0 + i * 4);
        s  += v[i].x + v[i].y + v[i].z + v[i].w;
        ss += v[i].x * v[i].x + v[i].y * v[i].y + v[i].z * v[i].z + v[i].w * v[i].w;
    }
    #pragma unroll
    for (int off = 16; off >= 1; off >>= 1) {
        s  += __shfl_xor(s, off, 32);
        ss += __shfl_xor(ss, off, 32);
    }
    float mean = s * (1.0f / DMODEL);
    float var  = ss * (1.0f / DMODEL) - mean * mean;
    float inv  = rsqrtf(var + 1e-5f);
    u16* orow = out + (size_t)tok * DMODEL;
    #pragma unroll
    for (int i = 0; i < 4; i++) {
        float4 vv = v[i];
        int d = d0 + i * 4;
        orow[d + 0] = f2bf((vv.x - mean) * inv * g[d + 0] + b[d + 0]);
        orow[d + 1] = f2bf((vv.y - mean) * inv * g[d + 1] + b[d + 1]);
        orow[d + 2] = f2bf((vv.z - mean) * inv * g[d + 2] + b[d + 2]);
        orow[d + 3] = f2bf((vv.w - mean) * inv * g[d + 3] + b[d + 3]);
    }
}

// ---------------------------------------------------------------------------
// Windowed attention, fp32. One wave per (b, head, q). Keys k with
// 0 <= q-k < 256 and !kpm[k]. Output o written in bf16 (B*TD, 512).
// ---------------------------------------------------------------------------
__global__ __launch_bounds__(256)
void attn_kernel(const float* __restrict__ qkv, const float* __restrict__ kpm,
                 u16* __restrict__ o)
{
    __shared__ float sq[8][DKH];
    __shared__ float sp[8][WINSZ];
    int w = threadIdx.x >> 5;
    int lane = threadIdx.x & 31;
    int blk = blockIdx.x;                 // BATCH*NHEAD*(TDN/8)
    int qt = blk & 127;
    int bh = blk >> 7;
    int hh = bh & (NHEAD - 1);
    int b  = bh >> 3;
    int q  = qt * 8 + w;
    size_t tokQ = (size_t)(b * TDN + q) * 1536;

    int d0 = lane * 2;
    float2 qv = *(const float2*)(qkv + tokQ + hh * DKH + d0);
    sq[w][d0] = qv.x; sq[w][d0 + 1] = qv.y;
    __syncthreads();

    int kbase = q - (WINSZ - 1);
    float s[8];
    float mmax = -INFINITY;
    #pragma unroll
    for (int i = 0; i < 8; i++) {
        int kk = i * 32 + lane;
        int kt = kbase + kk;
        float sc = -INFINITY;
        if (kt >= 0 && kpm[b * TDN + kt] == 0.0f) {
            const float* kp = qkv + (size_t)(b * TDN + kt) * 1536 + DMODEL + hh * DKH;
            float acc = 0.0f;
            #pragma unroll
            for (int d = 0; d < DKH; d += 4) {
                float4 kv4 = *(const float4*)(kp + d);
                acc += sq[w][d] * kv4.x + sq[w][d + 1] * kv4.y
                     + sq[w][d + 2] * kv4.z + sq[w][d + 3] * kv4.w;
            }
            sc = acc * 0.125f;            // 1/sqrt(64)
        }
        s[i] = sc;
        mmax = fmaxf(mmax, sc);
    }
    #pragma unroll
    for (int off = 16; off >= 1; off >>= 1) mmax = fmaxf(mmax, __shfl_xor(mmax, off, 32));
    float p[8];
    float lsum = 0.0f;
    #pragma unroll
    for (int i = 0; i < 8; i++) {
        p[i] = (s[i] == -INFINITY) ? 0.0f : __expf(s[i] - mmax);
        lsum += p[i];
    }
    #pragma unroll
    for (int off = 16; off >= 1; off >>= 1) lsum += __shfl_xor(lsum, off, 32);
    float inv = (lsum > 0.0f) ? 1.0f / lsum : 0.0f;
    #pragma unroll
    for (int i = 0; i < 8; i++) sp[w][i * 32 + lane] = p[i] * inv;
    __syncthreads();

    float a0 = 0.0f, a1 = 0.0f;
    int kk0 = (q >= WINSZ - 1) ? 0 : (WINSZ - 1 - q);
    for (int kk = kk0; kk < WINSZ; kk++) {
        float pw = sp[w][kk];
        int kt = kbase + kk;
        const float* vp = qkv + (size_t)(b * TDN + kt) * 1536 + 2 * DMODEL + hh * DKH + d0;
        float2 vv = *(const float2*)(vp);
        a0 += pw * vv.x;
        a1 += pw * vv.y;
    }
    u16* op = o + (size_t)(b * TDN + q) * DMODEL + hh * DKH + d0;
    op[0] = f2bf(a0);
    op[1] = f2bf(a1);
}

// ---------------------------------------------------------------------------
// Tiny head projections (N = 1 or 2): VALU dot product.
// mode: 0 = identity, 1 = sigmoid.
// ---------------------------------------------------------------------------
__global__ void head2_kernel(const u16* __restrict__ G, const float* __restrict__ W,
                             const float* __restrict__ bias, float* __restrict__ outp,
                             int C, int J, int mode)
{
    int idx = blockIdx.x * blockDim.x + threadIdx.x;
    if (idx >= NTOK * J) return;
    int t = idx / J;
    int j = idx - t * J;
    const u16* g = G + (size_t)t * C;
    float acc = bias[j];
    for (int c = 0; c < C; c++) acc += bf2f(g[c]) * W[(size_t)c * J + j];
    if (mode == 1) acc = 1.0f / (1.0f + __expf(-acc));
    outp[idx] = acc;
}

// ---------------------------------------------------------------------------
// Host launcher
// ---------------------------------------------------------------------------
#define OFF_H    ((size_t)0)              //  h fp32       : 4096*512*4  = 8,388,608
#define OFF_HN   ((size_t)8388608)        //  hn bf16      : 4096*512*2  = 4,194,304
#define OFF_QKV  ((size_t)12582912)       //  qkv fp32     : 4096*1536*4 = 25,165,824
#define OFF_O    ((size_t)37748736)       //  o bf16       : 4096*512*2  = 4,194,304
#define OFF_H1   ((size_t)41943040)       //  h1 bf16      : 4096*2048*2 = 16,777,216 (reused for head G1s)
#define OFF_XD   ((size_t)58720256)       //  xdown fp32   : 16,384
#define OFF_KPM  ((size_t)58736640)       //  kpm fp32     : 16,384
#define OFF_WSCR ((size_t)58753024)       //  bf16 weight scratch : 6,291,456 (reused per layer)

extern "C" void kernel_launch(void* const* d_in, const int* in_sizes, int n_in,
                              void* d_out, int out_size, void* d_ws, size_t ws_size,
                              hipStream_t stream)
{
    (void)out_size; (void)ws_size;
    // ---- identify inputs (insertion order vs jax-pytree alphabetical order)
    int I_x, I_convw, I_convb, I_inw, I_inb, I_pos, I_lnfg, I_lnfb;
    int I_binw1, I_binb1, I_binw2, I_binb2;
    int I_anow1, I_anob1, I_anow2, I_anob2;
    int I_cauw1, I_caub1, I_cauw2, I_caub2;
    int L_ln1g[4], L_ln1b[4], L_wqkv[4], L_wout[4], L_ln2g[4], L_ln2b[4];
    int L_w1[4], L_b1[4], L_w2[4], L_b2[4];

    bool insertion = (n_in > 0 && in_sizes[0] == BATCH * 5120);
    if (insertion) {
        I_x = 0; I_convw = 1; I_convb = 2; I_inw = 3; I_inb = 4; I_pos = 5;
        for (int l = 0; l < 4; l++) {
            int base = 6 + l * 10;
            L_ln1g[l] = base + 0; L_ln1b[l] = base + 1;
            L_wqkv[l] = base + 2; L_wout[l] = base + 3;
            L_ln2g[l] = base + 4; L_ln2b[l] = base + 5;
            L_w1[l] = base + 6;  L_b1[l] = base + 7;
            L_w2[l] = base + 8;  L_b2[l] = base + 9;
        }
        I_lnfg = 46; I_lnfb = 47;
        I_binw1 = 48; I_binb1 = 49; I_binw2 = 50; I_binb2 = 51;
        I_anow1 = 52; I_anob1 = 53; I_anow2 = 54; I_anob2 = 55;
        I_cauw1 = 56; I_caub1 = 57; I_cauw2 = 58; I_caub2 = 59;
    } else {
        // alphabetical pytree order: params subtree first, then x
        I_anob1 = 0; I_anob2 = 1; I_anow1 = 2; I_anow2 = 3;
        I_binb1 = 4; I_binb2 = 5; I_binw1 = 6; I_binw2 = 7;
        I_caub1 = 8; I_caub2 = 9; I_cauw1 = 10; I_cauw2 = 11;
        I_convb = 12; I_convw = 13; I_inb = 14; I_inw = 15;
        for (int l = 0; l < 4; l++) {
            int base = 16 + l * 10;
            L_b1[l] = base + 0;  L_b2[l] = base + 1;
            L_ln1b[l] = base + 2; L_ln1g[l] = base + 3;
            L_ln2b[l] = base + 4; L_ln2g[l] = base + 5;
            L_w1[l] = base + 6;  L_w2[l] = base + 7;
            L_wout[l] = base + 8; L_wqkv[l] = base + 9;
        }
        I_lnfb = 56; I_lnfg = 57; I_pos = 58; I_x = 59;
    }

    #define FP(i) ((const float*)d_in[(i)])
    char* ws = (char*)d_ws;
    float* h      = (float*)(ws + OFF_H);
    u16*   hn     = (u16*)  (ws + OFF_HN);
    float* qkv    = (float*)(ws + OFF_QKV);
    u16*   obuf   = (u16*)  (ws + OFF_O);
    u16*   h1     = (u16*)  (ws + OFF_H1);
    float* xdown  = (float*)(ws + OFF_XD);
    float* kpm    = (float*)(ws + OFF_KPM);
    u16*   wqkvT  = (u16*)  (ws + OFF_WSCR);
    u16*   woutT  = (u16*)  (ws + OFF_WSCR + 1572864);
    u16*   w1T    = (u16*)  (ws + OFF_WSCR + 2097152);
    u16*   w2T    = (u16*)  (ws + OFF_WSCR + 4194304);
    float* outF   = (float*)d_out;

    // ---- prologue
    downsample_kernel<<<(NTOK + 255) / 256, 256, 0, stream>>>(
        FP(I_x), FP(I_convw), FP(I_convb), xdown, kpm);
    embed_kernel<<<(NTOK * DMODEL + 255) / 256, 256, 0, stream>>>(
        xdown, FP(I_inw), FP(I_inb), FP(I_pos), h);

    // ---- transformer layers
    for (int l = 0; l < 4; l++) {
        convert_transpose_kernel<<<(DMODEL * 1536 + 255) / 256, 256, 0, stream>>>(
            FP(L_wqkv[l]), wqkvT, DMODEL, 1536);
        convert_transpose_kernel<<<(DMODEL * DMODEL + 255) / 256, 256, 0, stream>>>(
            FP(L_wout[l]), woutT, DMODEL, DMODEL);
        convert_transpose_kernel<<<(DMODEL * FFDIM + 255) / 256, 256, 0, stream>>>(
            FP(L_w1[l]), w1T, DMODEL, FFDIM);
        convert_transpose_kernel<<<(FFDIM * DMODEL + 255) / 256, 256, 0, stream>>>(
            FP(L_w2[l]), w2T, FFDIM, DMODEL);

        // hn = LN1(h) -> bf16
        layernorm_bf16<<<NTOK / 8, 256, 0, stream>>>(h, FP(L_ln1g[l]), FP(L_ln1b[l]), hn, NTOK);
        // qkv = hn @ w_qkv  (fp32 out)
        {
            int waves = (NTOK / 32) * (1536 / 32);
            gemm_bf16_wmma<0, 0, 0, 0><<<waves / 8, 256, 0, stream>>>(
                hn, wqkvT, nullptr, nullptr, qkv, nullptr, NTOK, 1536, DMODEL);
        }
        // attention -> o (bf16)
        attn_kernel<<<BATCH * NHEAD * (TDN / 8), 256, 0, stream>>>(qkv, kpm, obuf);
        // h = h + o @ w_out
        {
            int waves = (NTOK / 32) * (DMODEL / 32);
            gemm_bf16_wmma<0, 0, 1, 0><<<waves / 8, 256, 0, stream>>>(
                obuf, woutT, nullptr, h, h, nullptr, NTOK, DMODEL, DMODEL);
        }
        // hn = LN2(h) -> bf16
        layernorm_bf16<<<NTOK / 8, 256, 0, stream>>>(h, FP(L_ln2g[l]), FP(L_ln2b[l]), hn, NTOK);
        // h1 = gelu(hn @ w1 + b1) -> bf16
        {
            int waves = (NTOK / 32) * (FFDIM / 32);
            gemm_bf16_wmma<1, 1, 0, 1><<<waves / 8, 256, 0, stream>>>(
                hn, w1T, FP(L_b1[l]), nullptr, nullptr, h1, NTOK, FFDIM, DMODEL);
        }
        // h = h + h1 @ w2 + b2
        {
            int waves = (NTOK / 32) * (DMODEL / 32);
            gemm_bf16_wmma<1, 0, 1, 0><<<waves / 8, 256, 0, stream>>>(
                h1, w2T, FP(L_b2[l]), h, h, nullptr, NTOK, DMODEL, FFDIM);
        }
    }

    // ---- final LN + heads
    layernorm_bf16<<<NTOK / 8, 256, 0, stream>>>(h, FP(I_lnfg), FP(I_lnfb), hn, NTOK);

    u16* binw1T = (u16*)(ws + OFF_WSCR);
    u16* anow1T = (u16*)(ws + OFF_WSCR + 262144);
    u16* cauw1T = (u16*)(ws + OFF_WSCR + 393216);
    convert_transpose_kernel<<<(DMODEL * 256 + 255) / 256, 256, 0, stream>>>(
        FP(I_binw1), binw1T, DMODEL, 256);
    convert_transpose_kernel<<<(DMODEL * 128 + 255) / 256, 256, 0, stream>>>(
        FP(I_anow1), anow1T, DMODEL, 128);
    convert_transpose_kernel<<<(DMODEL * 128 + 255) / 256, 256, 0, stream>>>(
        FP(I_cauw1), cauw1T, DMODEL, 128);

    u16* bin_g1 = h1;                        // 4096*256 bf16
    u16* ano_g1 = h1 + (size_t)NTOK * 256;   // 4096*128 bf16
    u16* cau_g1 = ano_g1 + (size_t)NTOK * 128;

    {
        int waves = (NTOK / 32) * (256 / 32);
        gemm_bf16_wmma<1, 1, 0, 1><<<waves / 8, 256, 0, stream>>>(
            hn, binw1T, FP(I_binb1), nullptr, nullptr, bin_g1, NTOK, 256, DMODEL);
    }
    {
        int waves = (NTOK / 32) * (128 / 32);
        gemm_bf16_wmma<1, 1, 0, 1><<<waves / 8, 256, 0, stream>>>(
            hn, anow1T, FP(I_anob1), nullptr, nullptr, ano_g1, NTOK, 128, DMODEL);
        gemm_bf16_wmma<1, 1, 0, 1><<<waves / 8, 256, 0, stream>>>(
            hn, cauw1T, FP(I_caub1), nullptr, nullptr, cau_g1, NTOK, 128, DMODEL);
    }

    // output layout: binary (4096*2) | anomaly (4096) | caustic (4096)
    head2_kernel<<<(NTOK * 2 + 255) / 256, 256, 0, stream>>>(
        bin_g1, FP(I_binw2), FP(I_binb2), outF, 256, 2, 0);
    head2_kernel<<<(NTOK + 255) / 256, 256, 0, stream>>>(
        ano_g1, FP(I_anow2), FP(I_anob2), outF + NTOK * 2, 128, 1, 0);
    head2_kernel<<<(NTOK + 255) / 256, 256, 0, stream>>>(
        cau_g1, FP(I_cauw2), FP(I_caub2), outF + NTOK * 3, 128, 1, 1);
    #undef FP
}
